// Content_Attn_5875515261510
// MI455X (gfx1250) — matrix operand order repaired and checked by reading, not compiled, not run on hardware
//
#include <hip/hip_runtime.h>

// dims: B=32, S=F=512, T=1024
#define BATCH 32
#define SDIM  512
#define TDIM  1024

typedef __bf16 bf16;
typedef __attribute__((ext_vector_type(16))) __bf16 v16bf;
typedef __attribute__((ext_vector_type(8)))  __bf16 v8bf;
typedef __attribute__((ext_vector_type(4)))  __bf16 v4bf;
typedef __attribute__((ext_vector_type(8)))  float  v8f;

#define LDSW 40   // 32 bf16 payload + 8 pad -> 80B row stride, 16B aligned, conflict-free

// A fragment (16x32 MxK): lane m=lane&15, kg=lane>>4; v0-3 <- K=kg*8.., v4-7 <- K=16+kg*8..
__device__ __forceinline__ v16bf frag_a(const bf16* row, int kg) {
  union { v16bf v; v8bf h[2]; } u;
  u.h[0] = *(const v8bf*)(row + kg * 8);
  u.h[1] = *(const v8bf*)(row + 16 + kg * 8);
  return u.v;
}
// B fragment (32x16 KxN): lane n=lane&15, kh=lane>>4; v0-3 <- K=kh*8.., v4-7 <- K=16+kh*8..
__device__ __forceinline__ v16bf frag_b(const bf16* row, int kh) {
  union { v16bf v; v8bf h[2]; } u;
  u.h[0] = *(const v8bf*)(row + kh * 8);
  u.h[1] = *(const v8bf*)(row + 16 + kh * 8);
  return u.v;
}
__device__ __forceinline__ v4bf pack4(float4 v) {
  v4bf p; p[0] = (bf16)v.x; p[1] = (bf16)v.y; p[2] = (bf16)v.z; p[3] = (bf16)v.w;
  return p;
}
// transposed store: float4 along N -> 4 bf16 rows, same K column
__device__ __forceinline__ void store_tr(bf16* s, int nB, int kcol, float4 v) {
  s[(nB * 4 + 0) * LDSW + kcol] = (bf16)v.x;
  s[(nB * 4 + 1) * LDSW + kcol] = (bf16)v.y;
  s[(nB * 4 + 2) * LDSW + kcol] = (bf16)v.z;
  s[(nB * 4 + 3) * LDSW + kcol] = (bf16)v.w;
}

#define WMMA_BF16(A, Bm, Cacc) \
  __builtin_amdgcn_wmma_f32_16x16x32_bf16(false, (A), false, (Bm), (short)0, (Cacc), false, false)

// ---------------------------------------------------------------------------
// K1: Out[b] = W (512x512, row-major) @ Z[b] (512x1024, row-major)
// grid (TDIM/64, SDIM/64, BATCH), 256 threads (8 waves), double-buffered LDS
// ---------------------------------------------------------------------------
__global__ __launch_bounds__(256)
void k_gemm_w_z(const float* __restrict__ W, const float* __restrict__ Z,
                float* __restrict__ Out) {
  __shared__ __align__(16) bf16 sA[2][64 * LDSW];
  __shared__ __align__(16) bf16 sB[2][64 * LDSW];
  const int tid = threadIdx.x;
  const int b  = blockIdx.z;
  const int m0 = blockIdx.y * 64, n0 = blockIdx.x * 64;
  const float* Zb = Z + b * SDIM * TDIM;
  float*       Ob = Out + b * SDIM * TDIM;
  const int lane = tid & 31, l15 = lane & 15, kg = lane >> 4;
  const int w = tid >> 5, msub = w & 3, nhalf = w >> 2;
  const int rA = tid >> 3, cA = tid & 7;    // A: rows rA, rA+32, col grp cA
  const int kB = tid >> 4, nB = tid & 15;   // B: k rows kB, kB+16, n grp nB
  v8f acc0 = {}, acc1 = {};

  float4 a0, a1, bb0, bb1;
  // prologue: load+stage k0 = 0 into buffer 0
  a0  = *(const float4*)(W  + (m0 + rA)      * SDIM + cA * 4);
  a1  = *(const float4*)(W  + (m0 + rA + 32) * SDIM + cA * 4);
  bb0 = *(const float4*)(Zb + kB        * TDIM + n0 + nB * 4);
  bb1 = *(const float4*)(Zb + (kB + 16) * TDIM + n0 + nB * 4);
  *(v4bf*)(sA[0] + rA * LDSW + cA * 4)        = pack4(a0);
  *(v4bf*)(sA[0] + (rA + 32) * LDSW + cA * 4) = pack4(a1);
  store_tr(sB[0], nB, kB,      bb0);
  store_tr(sB[0], nB, kB + 16, bb1);
  __syncthreads();

  const int NK = SDIM / 32;
  for (int ki = 0; ki < NK; ++ki) {
    const int p = ki & 1;
    const bool nxt = (ki + 1) < NK;
    if (nxt) {                                  // issue next tile's global loads early
      int k0 = (ki + 1) * 32;
      a0  = *(const float4*)(W  + (m0 + rA)      * SDIM + k0 + cA * 4);
      a1  = *(const float4*)(W  + (m0 + rA + 32) * SDIM + k0 + cA * 4);
      bb0 = *(const float4*)(Zb + (k0 + kB)      * TDIM + n0 + nB * 4);
      bb1 = *(const float4*)(Zb + (k0 + kB + 16) * TDIM + n0 + nB * 4);
    }
    v16bf a  = frag_a(sA[p] + (msub * 16 + l15) * LDSW, kg);
    v16bf b0 = frag_b(sB[p] + (nhalf * 32 + l15) * LDSW, kg);
    v16bf b1 = frag_b(sB[p] + (nhalf * 32 + 16 + l15) * LDSW, kg);
    acc0 = WMMA_BF16(a, b0, acc0);
    acc1 = WMMA_BF16(a, b1, acc1);
    if (nxt) {                                  // stage into the other buffer
      const int q = p ^ 1;
      *(v4bf*)(sA[q] + rA * LDSW + cA * 4)        = pack4(a0);
      *(v4bf*)(sA[q] + (rA + 32) * LDSW + cA * 4) = pack4(a1);
      store_tr(sB[q], nB, kB,      bb0);
      store_tr(sB[q], nB, kB + 16, bb1);
    }
    __syncthreads();                            // single barrier per K-step
  }
  #pragma unroll
  for (int i = 0; i < 8; ++i) {
    int m = m0 + msub * 16 + kg * 8 + i;
    int n = n0 + nhalf * 32 + l15;
    Ob[m * TDIM + n]      = acc0[i];
    Ob[m * TDIM + n + 16] = acc1[i];
  }
}

// ---------------------------------------------------------------------------
// K2: C[b] = tanh( z_r[b]^T (TxS) @ aux[b] (SxT) )   M=N=1024, K=512
// grid (16, 16, BATCH), double-buffered
// ---------------------------------------------------------------------------
__global__ __launch_bounds__(256)
void k_gemm_C(const float* __restrict__ Zr, const float* __restrict__ Aux,
              float* __restrict__ Cout) {
  __shared__ __align__(16) bf16 sA[2][64 * LDSW];
  __shared__ __align__(16) bf16 sB[2][64 * LDSW];
  const int tid = threadIdx.x;
  const int b  = blockIdx.z;
  const int m0 = blockIdx.y * 64, n0 = blockIdx.x * 64;   // m0: t, n0: h
  const float* Zb = Zr  + b * SDIM * TDIM;
  const float* Ab = Aux + b * SDIM * TDIM;
  float*       Cb = Cout + b * TDIM * TDIM;
  const int lane = tid & 31, l15 = lane & 15, kg = lane >> 4;
  const int w = tid >> 5, msub = w & 3, nhalf = w >> 2;
  const int kB = tid >> 4, nB = tid & 15;   // both A and B staged transposed
  v8f acc0 = {}, acc1 = {};

  float4 a0, a1, bb0, bb1;
  a0  = *(const float4*)(Zb + kB        * TDIM + m0 + nB * 4);
  a1  = *(const float4*)(Zb + (kB + 16) * TDIM + m0 + nB * 4);
  bb0 = *(const float4*)(Ab + kB        * TDIM + n0 + nB * 4);
  bb1 = *(const float4*)(Ab + (kB + 16) * TDIM + n0 + nB * 4);
  store_tr(sA[0], nB, kB,      a0);
  store_tr(sA[0], nB, kB + 16, a1);
  store_tr(sB[0], nB, kB,      bb0);
  store_tr(sB[0], nB, kB + 16, bb1);
  __syncthreads();

  const int NK = SDIM / 32;
  for (int ki = 0; ki < NK; ++ki) {
    const int p = ki & 1;
    const bool nxt = (ki + 1) < NK;
    if (nxt) {
      int k0 = (ki + 1) * 32;
      a0  = *(const float4*)(Zb + (k0 + kB)      * TDIM + m0 + nB * 4);
      a1  = *(const float4*)(Zb + (k0 + kB + 16) * TDIM + m0 + nB * 4);
      bb0 = *(const float4*)(Ab + (k0 + kB)      * TDIM + n0 + nB * 4);
      bb1 = *(const float4*)(Ab + (k0 + kB + 16) * TDIM + n0 + nB * 4);
    }
    v16bf a  = frag_a(sA[p] + (msub * 16 + l15) * LDSW, kg);
    v16bf b0 = frag_b(sB[p] + (nhalf * 32 + l15) * LDSW, kg);
    v16bf b1 = frag_b(sB[p] + (nhalf * 32 + 16 + l15) * LDSW, kg);
    acc0 = WMMA_BF16(a, b0, acc0);
    acc1 = WMMA_BF16(a, b1, acc1);
    if (nxt) {
      const int q = p ^ 1;
      store_tr(sA[q], nB, kB,      a0);
      store_tr(sA[q], nB, kB + 16, a1);
      store_tr(sB[q], nB, kB,      bb0);
      store_tr(sB[q], nB, kB + 16, bb1);
    }
    __syncthreads();
  }
  #pragma unroll
  for (int i = 0; i < 8; ++i) {
    int m = m0 + msub * 16 + kg * 8 + i;
    int n = n0 + nhalf * 32 + l15;
    Cb[m * TDIM + n]      = tanhf(acc0[i]);
    Cb[m * TDIM + n + 16] = tanhf(acc1[i]);
  }
}

// ---------------------------------------------------------------------------
// K3: fused H_q + logits, double-buffered 48-step K loop:
//   Hacc[f,t] = sum_h aux2[f,h]*C[t,h] + sum_s W_q[f,s]*z_r[s,t]
//   logits[b,t] = sum_f w_hq[f]*tanh(Hacc[f,t])   (H_q never materialized)
// grid (TDIM/64, BATCH)
// ---------------------------------------------------------------------------
__global__ __launch_bounds__(256)
void k_Hq_logits(const float* __restrict__ Aux2, const float* __restrict__ Cmat,
                 const float* __restrict__ Wq, const float* __restrict__ Zr,
                 const float* __restrict__ whq, float* __restrict__ logits) {
  __shared__ __align__(16) bf16 sA[2][32 * LDSW];
  __shared__ __align__(16) bf16 sB[2][64 * LDSW];
  __shared__ float sW[SDIM];
  __shared__ float sLog[64];
  const int tid = threadIdx.x;
  const int b  = blockIdx.y;
  const int t0 = blockIdx.x * 64;
  const float* A2 = Aux2 + b * SDIM * TDIM;
  const float* Cb = Cmat + b * TDIM * TDIM;
  const float* Zb = Zr   + b * SDIM * TDIM;
  sW[tid] = whq[tid];
  sW[tid + 256] = whq[tid + 256];
  if (tid < 64) sLog[tid] = 0.0f;
  __syncthreads();
  const int lane = tid & 31, l15 = lane & 15, kg = lane >> 4;
  const int w = tid >> 5, fl = w >> 2, tsub = w & 3;
  const int rA = tid >> 3, cA = tid & 7;    // A (32 rows) + direct-B (rows rA, rA+32)
  const int kB = tid >> 4, nB = tid & 15;   // transposed-B

  for (int fp = 0; fp < 16; ++fp) {
    const int f0 = fp * 32;
    v8f acc = {};
    float4 la, lb0, lb1;
    // prologue: stage ks = 0 (h-phase) into buffer 0
    la  = *(const float4*)(A2 + (f0 + rA)      * TDIM + cA * 4);
    lb0 = *(const float4*)(Cb + (t0 + rA)      * TDIM + cA * 4);
    lb1 = *(const float4*)(Cb + (t0 + rA + 32) * TDIM + cA * 4);
    *(v4bf*)(sA[0] + rA * LDSW + cA * 4)        = pack4(la);
    *(v4bf*)(sB[0] + rA * LDSW + cA * 4)        = pack4(lb0);
    *(v4bf*)(sB[0] + (rA + 32) * LDSW + cA * 4) = pack4(lb1);
    __syncthreads();

    for (int ks = 0; ks < 48; ++ks) {
      const int p = ks & 1;
      const bool nxt = (ks + 1) < 48;
      const int kn = ks + 1;
      if (nxt) {
        if (kn < 32) {                          // next is h-phase: A=aux2, B=C (both direct)
          int k0 = kn * 32;
          la  = *(const float4*)(A2 + (f0 + rA)      * TDIM + k0 + cA * 4);
          lb0 = *(const float4*)(Cb + (t0 + rA)      * TDIM + k0 + cA * 4);
          lb1 = *(const float4*)(Cb + (t0 + rA + 32) * TDIM + k0 + cA * 4);
        } else {                                // next is s-phase: A=W_q direct, B=z_r transposed
          int k0 = (kn - 32) * 32;
          la  = *(const float4*)(Wq + (f0 + rA)      * SDIM + k0 + cA * 4);
          lb0 = *(const float4*)(Zb + (k0 + kB)      * TDIM + t0 + nB * 4);
          lb1 = *(const float4*)(Zb + (k0 + kB + 16) * TDIM + t0 + nB * 4);
        }
      }
      v16bf a  = frag_a(sA[p] + (fl * 16 + l15) * LDSW, kg);
      v16bf bb = frag_b(sB[p] + (tsub * 16 + l15) * LDSW, kg);
      acc = WMMA_BF16(a, bb, acc);
      if (nxt) {
        const int q = p ^ 1;
        *(v4bf*)(sA[q] + rA * LDSW + cA * 4) = pack4(la);
        if (kn < 32) {
          *(v4bf*)(sB[q] + rA * LDSW + cA * 4)        = pack4(lb0);
          *(v4bf*)(sB[q] + (rA + 32) * LDSW + cA * 4) = pack4(lb1);
        } else {
          store_tr(sB[q], nB, kB,      lb0);
          store_tr(sB[q], nB, kB + 16, lb1);
        }
      }
      __syncthreads();
    }
    float pl = 0.0f;
    #pragma unroll
    for (int i = 0; i < 8; ++i) {
      int f = f0 + fl * 16 + kg * 8 + i;
      pl += sW[f] * tanhf(acc[i]);
    }
    atomicAdd(&sLog[tsub * 16 + l15], pl);   // ds_add_f32
  }
  __syncthreads();
  if (tid < 64) logits[b * TDIM + t0 + tid] = sLog[tid];
}

// ---------------------------------------------------------------------------
// K4: softmax over t, then attn[b,s] = sum_t z_r[b,s,t] * a_q[b,t].  grid(BATCH)
// ---------------------------------------------------------------------------
__global__ __launch_bounds__(256)
void k_softmax_attn(const float* __restrict__ logits, const float* __restrict__ Zr,
                    float* __restrict__ aq_out, float* __restrict__ attn_out) {
  __shared__ float sAq[TDIM];
  __shared__ float sRed[256];
  const int tid = threadIdx.x, b = blockIdx.x;
  const float* lg = logits + b * TDIM;
  float4 v = *(const float4*)(lg + tid * 4);
  float mx = fmaxf(fmaxf(v.x, v.y), fmaxf(v.z, v.w));
  sRed[tid] = mx; __syncthreads();
  for (int s = 128; s > 0; s >>= 1) {
    if (tid < s) sRed[tid] = fmaxf(sRed[tid], sRed[tid + s]);
    __syncthreads();
  }
  mx = sRed[0];
  __syncthreads();
  float4 e;
  e.x = expf(v.x - mx); e.y = expf(v.y - mx); e.z = expf(v.z - mx); e.w = expf(v.w - mx);
  sRed[tid] = e.x + e.y + e.z + e.w; __syncthreads();
  for (int s = 128; s > 0; s >>= 1) {
    if (tid < s) sRed[tid] += sRed[tid + s];
    __syncthreads();
  }
  float inv = 1.0f / sRed[0];
  float4 a; a.x = e.x * inv; a.y = e.y * inv; a.z = e.z * inv; a.w = e.w * inv;
  sAq[tid * 4 + 0] = a.x; sAq[tid * 4 + 1] = a.y; sAq[tid * 4 + 2] = a.z; sAq[tid * 4 + 3] = a.w;
  *(float4*)(aq_out + b * TDIM + tid * 4) = a;
  __syncthreads();
  #pragma unroll
  for (int rr = 0; rr < 2; ++rr) {
    int s = tid + rr * 256;
    const float* zrow = Zr + b * SDIM * TDIM + s * TDIM;
    float acc = 0.0f;
    for (int t = 0; t < TDIM; t += 4) {
      float4 z = *(const float4*)(zrow + t);
      acc += z.x * sAq[t] + z.y * sAq[t + 1] + z.z * sAq[t + 2] + z.w * sAq[t + 3];
    }
    attn_out[b * SDIM + s] = acc;
  }
}

// ---------------------------------------------------------------------------
extern "C" void kernel_launch(void* const* d_in, const int* in_sizes, int n_in,
                              void* d_out, int out_size, void* d_ws, size_t ws_size,
                              hipStream_t stream) {
  const float* z_robot = (const float*)d_in[0];
  const float* z_human = (const float*)d_in[1];
  const float* W_b     = (const float*)d_in[2];
  const float* W_q     = (const float*)d_in[3];
  const float* W_v     = (const float*)d_in[4];
  const float* w_hq    = (const float*)d_in[5];

  float* out  = (float*)d_out;
  float* attn = out;                                   // [B,S,1]  = 16384
  float* aq   = out + BATCH * SDIM;                    // [B,1,T]  = 32768
  float* Cout = out + BATCH * SDIM + BATCH * TDIM;     // [B,T,T]

  float* buf    = (float*)d_ws;                        // [B,S,T] : aux then aux2 (reused)
  float* logits = buf + BATCH * SDIM * TDIM;           // [B,T]

  dim3 blk(256);
  // aux = W_b @ z_human
  k_gemm_w_z<<<dim3(TDIM / 64, SDIM / 64, BATCH), blk, 0, stream>>>(W_b, z_human, buf);
  // C = tanh(z_r^T @ aux)
  k_gemm_C<<<dim3(TDIM / 64, TDIM / 64, BATCH), blk, 0, stream>>>(z_robot, buf, Cout);
  // aux2 = W_v @ z_human   (reuses buf; aux is dead)
  k_gemm_w_z<<<dim3(TDIM / 64, SDIM / 64, BATCH), blk, 0, stream>>>(W_v, z_human, buf);
  // logits = w_hq . tanh(W_q@z_r + aux2@C^T)   (H_q never materialized)
  k_Hq_logits<<<dim3(TDIM / 64, BATCH), blk, 0, stream>>>(buf, Cout, W_q, z_robot, w_hq, logits);
  // a_q = softmax(logits); attn = z_r @ a_q^T
  k_softmax_attn<<<BATCH, blk, 0, stream>>>(logits, z_robot, aq, attn);
}
